// MeshGraphNet_18537078850012
// MI455X (gfx1250) — matrix operand order repaired and checked by reading, not compiled, not run on hardware
//
#include <hip/hip_runtime.h>
#include <hip/hip_bf16.h>
#include <cstdint>

// ---------------------------------------------------------------------------
// MeshGraphNet forward for MI455X (gfx1250, wave32, WMMA).
// All dense math uses v_wmma_f32_16x16x32_f16 (f16 inputs, f32 accumulate).
// f32 master state for residual/LN numerics, f16 mirrors for gathers/GEMMs.
// 32-row M-tiles per 128-thread block: each B fragment feeds 2 WMMAs.
// ---------------------------------------------------------------------------

typedef __attribute__((ext_vector_type(16))) _Float16 v16h;
typedef __attribute__((ext_vector_type(8)))  _Float16 v8h;
typedef __attribute__((ext_vector_type(8)))  float    v8f;

#define HIDN 128
#define ROWS 32            // M rows per block (2 x 16-row WMMA tiles)

union H4 { _Float16 h[4]; uint32_t u[2]; };

__device__ __forceinline__ v8f wmma16(v16h a, v16h b, v8f c) {
  // 8 args: (neg_a, A, neg_b, B, c_mod, C, reuse_a, reuse_b)
  return __builtin_amdgcn_wmma_f32_16x16x32_f16(false, a, false, b, (short)0, c,
                                                false, false);
}

// ROWS x Kdim @ Kdim x (NT*16) GEMM for one 128-thread block (4 wave32).
// sA: LDS row-major ROWS x Kdim f16.  Bp: packed B fragments in global:
//   frag (nt*KT+kt): 32 lanes x 16 halves, lane%16 = N-in-tile, lane/16 + halves = K-in-tile.
// Output: f32 written to LDS, row-major ROWS x outStride.
__device__ __forceinline__ void gemm_tile(const _Float16* sA, int Kdim,
                                          const _Float16* __restrict__ Bp,
                                          const float* __restrict__ bias,
                                          bool relu, float* sOut,
                                          int NT, int outStride) {
  const int lane = threadIdx.x & 31;
  const int wave = threadIdx.x >> 5;
  const int row  = lane & 15;                 // A row-in-tile / D column-in-tile
  const int koff = (lane < 16) ? 0 : 8;       // A fragment K sub-offset
  const int mo   = (lane < 16) ? 0 : 8;       // D fragment M offset
  const int KT   = Kdim >> 5;
  const v8f vzero = {0.f, 0.f, 0.f, 0.f, 0.f, 0.f, 0.f, 0.f};
  for (int nt = wave; nt < NT; nt += 4) {
    v8f acc0 = vzero, acc1 = vzero;
    for (int kt = 0; kt < KT; ++kt) {
      // one B fragment shared by both 16-row A tiles
      const v16h b = *(const v16h*)(Bp + (((size_t)(nt * KT + kt)) * 32 + lane) * 16);
      // A 16x32 f16 fragment (ISA 7.12.2): lanes 0-15 hold K 0..7 / 16..23,
      // lanes 16-31 hold K 8..15 / 24..31 -> two contiguous 16B LDS loads.
      const _Float16* pa0 = sA + (size_t)row * Kdim + kt * 32 + koff;
      const _Float16* pa1 = pa0 + (size_t)16 * Kdim;
      v8h lo0 = *(const v8h*)pa0, hi0 = *(const v8h*)(pa0 + 16);
      v8h lo1 = *(const v8h*)pa1, hi1 = *(const v8h*)(pa1 + 16);
      v16h a0 = __builtin_shufflevector(lo0, hi0, 0,1,2,3,4,5,6,7,8,9,10,11,12,13,14,15);
      v16h a1 = __builtin_shufflevector(lo1, hi1, 0,1,2,3,4,5,6,7,8,9,10,11,12,13,14,15);
      acc0 = wmma16(a0, b, acc0);
      acc1 = wmma16(a1, b, acc1);
    }
    const int col = nt * 16 + row;
    const float bv = bias ? bias[col] : 0.0f;
#pragma unroll
    for (int v = 0; v < 8; ++v) {
      float x0 = acc0[v] + bv;
      float x1 = acc1[v] + bv;
      if (relu) { x0 = x0 > 0.f ? x0 : 0.f; x1 = x1 > 0.f ? x1 : 0.f; }
      sOut[(v + mo) * outStride + col]      = x0;
      sOut[(16 + v + mo) * outStride + col] = x1;
    }
  }
}

// Per-row LayerNorm stats over 128 columns of a ROWS x 128 f32 LDS tile.
__device__ __forceinline__ void ln_stats(const float* sOut, float* sMu, float* sRs) {
  if (threadIdx.x < ROWS) {
    const int r = threadIdx.x;
    float s = 0.f, s2 = 0.f;
    for (int c = 0; c < HIDN; ++c) { float v = sOut[r * HIDN + c]; s += v; s2 += v * v; }
    const float mu  = s * (1.0f / HIDN);
    const float var = s2 * (1.0f / HIDN) - mu * mu;
    sMu[r] = mu;
    sRs[r] = rsqrtf(var + 1e-5f);
  }
}

// f32 LDS tile -> f16 LDS tile, 4 contiguous elements per thread per step.
__device__ __forceinline__ void cvt_tile(const float* sOut, _Float16* sH) {
  for (int q = threadIdx.x; q < ROWS * HIDN / 4; q += 128) {
    const float4 f = *(const float4*)(sOut + q * 4);
    H4 h;
    h.h[0] = (_Float16)f.x; h.h[1] = (_Float16)f.y;
    h.h[2] = (_Float16)f.z; h.h[3] = (_Float16)f.w;
    *(H4*)(sH + q * 4) = h;
  }
}

// ---------------------------------------------------------------------------
// Weight packer: W (Kreal x Nreal, f32 row-major) -> per-lane f16 B fragments.
// grid = NT*KT blocks of 32 threads; frag = blockIdx.x = nt*KT + kt.
// ---------------------------------------------------------------------------
__global__ void k_pack_w(const float* __restrict__ W, int Kreal, int Kpad,
                         int Nreal, _Float16* __restrict__ out) {
  const int KT   = Kpad >> 5;
  const int frag = blockIdx.x;
  const int kt   = frag % KT;
  const int nt   = frag / KT;
  const int lane = threadIdx.x;      // lane -> N within tile, elements -> K (table order)
  const int n    = nt * 16 + (lane & 15);
  _Float16* o = out + ((size_t)frag * 32 + lane) * 16;
#pragma unroll
  for (int e = 0; e < 16; ++e) {
    const int k = kt * 32 + ((e & 8) << 1) + (lane >> 4) * 8 + (e & 7);
    float v = 0.0f;
    if (k < Kreal && n < Nreal) v = W[(size_t)k * Nreal + n];
    o[e] = (_Float16)v;
  }
}

// ---------------------------------------------------------------------------
// Node encoder: x_norm(11) -> 128 -> relu -> 128 -> LN.  Also zero agg.
// ---------------------------------------------------------------------------
__global__ __launch_bounds__(128)
void k_node_enc(const float* __restrict__ x, const float* __restrict__ meanv,
                const float* __restrict__ stdv,
                const _Float16* __restrict__ W1p, const float* __restrict__ b1,
                const _Float16* __restrict__ W2p, const float* __restrict__ b2,
                const float* __restrict__ g, const float* __restrict__ be,
                float* __restrict__ xf32, _Float16* __restrict__ xf16,
                float* __restrict__ agg) {
  __shared__ __align__(16) _Float16 sA[ROWS * 32];
  __shared__ __align__(16) _Float16 sH[ROWS * HIDN];
  __shared__ __align__(16) float sOut[ROWS * HIDN];
  __shared__ float sMu[ROWS], sRs[ROWS];
  const int n0 = blockIdx.x * ROWS;
  for (int idx = threadIdx.x; idx < ROWS * 32; idx += 128) {
    const int r = idx >> 5, k = idx & 31;
    float v = 0.0f;
    if (k < 11) v = (x[(size_t)(n0 + r) * 11 + k] - meanv[k]) / stdv[k];
    sA[idx] = (_Float16)v;
  }
  __syncthreads();
  gemm_tile(sA, 32, W1p, b1, true, sOut, 8, HIDN);
  __syncthreads();
  cvt_tile(sOut, sH);
  __syncthreads();
  gemm_tile(sH, HIDN, W2p, b2, false, sOut, 8, HIDN);
  __syncthreads();
  ln_stats(sOut, sMu, sRs);
  __syncthreads();
  const float4 z4 = {0.f, 0.f, 0.f, 0.f};
  for (int q = threadIdx.x; q < ROWS * 32; q += 128) {   // 4 cols per step
    const int r = q >> 5, c = (q & 31) * 4;
    const float4 t = *(const float4*)(sOut + r * HIDN + c);
    const float4 gc = *(const float4*)(g + c);
    const float4 bc = *(const float4*)(be + c);
    float4 v;
    v.x = (t.x - sMu[r]) * sRs[r] * gc.x + bc.x;
    v.y = (t.y - sMu[r]) * sRs[r] * gc.y + bc.y;
    v.z = (t.z - sMu[r]) * sRs[r] * gc.z + bc.z;
    v.w = (t.w - sMu[r]) * sRs[r] * gc.w + bc.w;
    const size_t o = (size_t)(n0 + r) * HIDN + c;
    *(float4*)(xf32 + o) = v;
    H4 h;
    h.h[0] = (_Float16)v.x; h.h[1] = (_Float16)v.y;
    h.h[2] = (_Float16)v.z; h.h[3] = (_Float16)v.w;
    *(H4*)(xf16 + o) = h;
    *(float4*)(agg + o) = z4;   // initialize scatter accumulator for layer 1
  }
}

// ---------------------------------------------------------------------------
// Edge encoder: edge_attr_norm(3) -> 128 -> relu -> 128 -> LN.
// ---------------------------------------------------------------------------
__global__ __launch_bounds__(128)
void k_edge_enc(const float* __restrict__ ea, const float* __restrict__ meanv,
                const float* __restrict__ stdv,
                const _Float16* __restrict__ W1p, const float* __restrict__ b1,
                const _Float16* __restrict__ W2p, const float* __restrict__ b2,
                const float* __restrict__ g, const float* __restrict__ be,
                float* __restrict__ ef32, _Float16* __restrict__ ef16) {
  __shared__ __align__(16) _Float16 sA[ROWS * 32];
  __shared__ __align__(16) _Float16 sH[ROWS * HIDN];
  __shared__ __align__(16) float sOut[ROWS * HIDN];
  __shared__ float sMu[ROWS], sRs[ROWS];
  const int e0 = blockIdx.x * ROWS;
  for (int idx = threadIdx.x; idx < ROWS * 32; idx += 128) {
    const int r = idx >> 5, k = idx & 31;
    float v = 0.0f;
    if (k < 3) v = (ea[(size_t)(e0 + r) * 3 + k] - meanv[k]) / stdv[k];
    sA[idx] = (_Float16)v;
  }
  __syncthreads();
  gemm_tile(sA, 32, W1p, b1, true, sOut, 8, HIDN);
  __syncthreads();
  cvt_tile(sOut, sH);
  __syncthreads();
  gemm_tile(sH, HIDN, W2p, b2, false, sOut, 8, HIDN);
  __syncthreads();
  ln_stats(sOut, sMu, sRs);
  __syncthreads();
  for (int q = threadIdx.x; q < ROWS * 32; q += 128) {
    const int r = q >> 5, c = (q & 31) * 4;
    const float4 t = *(const float4*)(sOut + r * HIDN + c);
    const float4 gc = *(const float4*)(g + c);
    const float4 bc = *(const float4*)(be + c);
    float4 v;
    v.x = (t.x - sMu[r]) * sRs[r] * gc.x + bc.x;
    v.y = (t.y - sMu[r]) * sRs[r] * gc.y + bc.y;
    v.z = (t.z - sMu[r]) * sRs[r] * gc.z + bc.z;
    v.w = (t.w - sMu[r]) * sRs[r] * gc.w + bc.w;
    const size_t o = (size_t)(e0 + r) * HIDN + c;
    *(float4*)(ef32 + o) = v;
    H4 h;
    h.h[0] = (_Float16)v.x; h.h[1] = (_Float16)v.y;
    h.h[2] = (_Float16)v.z; h.h[3] = (_Float16)v.w;
    *(H4*)(ef16 + o) = h;
  }
}

// ---------------------------------------------------------------------------
// Processor edge update: ue = LN(MLP([x_dst | x_src | e])) + e; scatter by src.
// ---------------------------------------------------------------------------
__global__ __launch_bounds__(128)
void k_edge_proc(const int* __restrict__ srcIdx, const int* __restrict__ dstIdx,
                 const _Float16* __restrict__ xf16,
                 float* __restrict__ ef32, _Float16* __restrict__ ef16,
                 const _Float16* __restrict__ W1p, const float* __restrict__ b1,
                 const _Float16* __restrict__ W2p, const float* __restrict__ b2,
                 const float* __restrict__ g, const float* __restrict__ be,
                 float* __restrict__ agg) {
  __shared__ __align__(16) _Float16 sA[ROWS * 384];
  __shared__ __align__(16) _Float16 sH[ROWS * HIDN];
  __shared__ __align__(16) float sOut[ROWS * HIDN];
  __shared__ float sMu[ROWS], sRs[ROWS];
  __shared__ int sDst[ROWS], sSrc[ROWS];
  const int e0 = blockIdx.x * ROWS;
  if (threadIdx.x < ROWS) {
    sDst[threadIdx.x] = dstIdx[e0 + threadIdx.x];
    sSrc[threadIdx.x] = srcIdx[e0 + threadIdx.x];
  }
  __syncthreads();
  // Gather A = [x_i(dst) | x_j(src) | e] : ROWS x 384 f16, 16B chunks.
  for (int gq = threadIdx.x; gq < ROWS * 48; gq += 128) {
    const int r = gq / 48, c8 = gq % 48;
    const int seg = c8 >> 4, off = (c8 & 15) * 8;
    const _Float16* p;
    if (seg == 0)      p = xf16 + (size_t)sDst[r] * HIDN + off;
    else if (seg == 1) p = xf16 + (size_t)sSrc[r] * HIDN + off;
    else               p = ef16 + (size_t)(e0 + r) * HIDN + off;
    *(uint4*)(sA + r * 384 + seg * HIDN + off) = *(const uint4*)p;
  }
  __syncthreads();
  gemm_tile(sA, 384, W1p, b1, true, sOut, 8, HIDN);   // 12 K-tiles per n-tile
  __syncthreads();
  cvt_tile(sOut, sH);
  __syncthreads();
  gemm_tile(sH, HIDN, W2p, b2, false, sOut, 8, HIDN);
  __syncthreads();
  ln_stats(sOut, sMu, sRs);
  __syncthreads();
  for (int q = threadIdx.x; q < ROWS * 32; q += 128) {
    const int r = q >> 5, c = (q & 31) * 4;
    const float4 t = *(const float4*)(sOut + r * HIDN + c);
    const float4 gc = *(const float4*)(g + c);
    const float4 bc = *(const float4*)(be + c);
    const size_t o = (size_t)(e0 + r) * HIDN + c;
    const float4 eo = *(const float4*)(ef32 + o);    // residual
    float4 u;
    u.x = (t.x - sMu[r]) * sRs[r] * gc.x + bc.x + eo.x;
    u.y = (t.y - sMu[r]) * sRs[r] * gc.y + bc.y + eo.y;
    u.z = (t.z - sMu[r]) * sRs[r] * gc.z + bc.z + eo.z;
    u.w = (t.w - sMu[r]) * sRs[r] * gc.w + bc.w + eo.w;
    *(float4*)(ef32 + o) = u;
    H4 h;
    h.h[0] = (_Float16)u.x; h.h[1] = (_Float16)u.y;
    h.h[2] = (_Float16)u.z; h.h[3] = (_Float16)u.w;
    *(H4*)(ef16 + o) = h;
    float* ag = agg + (size_t)sSrc[r] * HIDN + c;    // segment_sum by src
    atomicAdd(ag + 0, u.x);
    atomicAdd(ag + 1, u.y);
    atomicAdd(ag + 2, u.z);
    atomicAdd(ag + 3, u.w);
  }
}

// ---------------------------------------------------------------------------
// Processor node update: x += LN(MLP([x | agg])); re-zero agg for next layer.
// ---------------------------------------------------------------------------
__global__ __launch_bounds__(128)
void k_node_proc(float* __restrict__ xf32, _Float16* __restrict__ xf16,
                 float* __restrict__ agg,
                 const _Float16* __restrict__ W1p, const float* __restrict__ b1,
                 const _Float16* __restrict__ W2p, const float* __restrict__ b2,
                 const float* __restrict__ g, const float* __restrict__ be) {
  __shared__ __align__(16) _Float16 sA[ROWS * 256];
  __shared__ __align__(16) _Float16 sH[ROWS * HIDN];
  __shared__ __align__(16) float sOut[ROWS * HIDN];
  __shared__ float sMu[ROWS], sRs[ROWS];
  const int n0 = blockIdx.x * ROWS;
  // cols 0..127 <- x (f16 copy, 16B chunks)
  for (int q = threadIdx.x; q < ROWS * 16; q += 128) {
    const int r = q >> 4, off = (q & 15) * 8;
    *(uint4*)(sA + r * 256 + off) = *(const uint4*)(xf16 + (size_t)(n0 + r) * HIDN + off);
  }
  // cols 128..255 <- agg (f32 -> f16)
  for (int q = threadIdx.x; q < ROWS * 32; q += 128) {
    const int r = q >> 5, c = (q & 31) * 4;
    const float4 a4 = *(const float4*)(agg + (size_t)(n0 + r) * HIDN + c);
    H4 h;
    h.h[0] = (_Float16)a4.x; h.h[1] = (_Float16)a4.y;
    h.h[2] = (_Float16)a4.z; h.h[3] = (_Float16)a4.w;
    *(H4*)(sA + r * 256 + HIDN + c) = h;
  }
  __syncthreads();
  // All agg reads for this block's rows done -> zero for next layer.
  const float4 z4 = {0.f, 0.f, 0.f, 0.f};
  for (int q = threadIdx.x; q < ROWS * 32; q += 128)
    *(float4*)(agg + (size_t)n0 * HIDN + q * 4) = z4;
  gemm_tile(sA, 256, W1p, b1, true, sOut, 8, HIDN);
  __syncthreads();
  cvt_tile(sOut, sH);
  __syncthreads();
  gemm_tile(sH, HIDN, W2p, b2, false, sOut, 8, HIDN);
  __syncthreads();
  ln_stats(sOut, sMu, sRs);
  __syncthreads();
  for (int q = threadIdx.x; q < ROWS * 32; q += 128) {
    const int r = q >> 5, c = (q & 31) * 4;
    const float4 t = *(const float4*)(sOut + r * HIDN + c);
    const float4 gc = *(const float4*)(g + c);
    const float4 bc = *(const float4*)(be + c);
    const size_t o = (size_t)(n0 + r) * HIDN + c;
    const float4 xo = *(const float4*)(xf32 + o);    // residual
    float4 xn;
    xn.x = (t.x - sMu[r]) * sRs[r] * gc.x + bc.x + xo.x;
    xn.y = (t.y - sMu[r]) * sRs[r] * gc.y + bc.y + xo.y;
    xn.z = (t.z - sMu[r]) * sRs[r] * gc.z + bc.z + xo.z;
    xn.w = (t.w - sMu[r]) * sRs[r] * gc.w + bc.w + xo.w;
    *(float4*)(xf32 + o) = xn;
    H4 h;
    h.h[0] = (_Float16)xn.x; h.h[1] = (_Float16)xn.y;
    h.h[2] = (_Float16)xn.z; h.h[3] = (_Float16)xn.w;
    *(H4*)(xf16 + o) = h;
  }
}

// ---------------------------------------------------------------------------
// Decoder: relu(x W1 + b1) W2 + b2 -> [N, 2] f32.
// ---------------------------------------------------------------------------
__global__ __launch_bounds__(128)
void k_decoder(const _Float16* __restrict__ xf16,
               const _Float16* __restrict__ W1p, const float* __restrict__ b1,
               const _Float16* __restrict__ W2p, const float* __restrict__ b2,
               float* __restrict__ out) {
  __shared__ __align__(16) _Float16 sA[ROWS * HIDN];
  __shared__ __align__(16) _Float16 sH[ROWS * HIDN];
  __shared__ __align__(16) float sOut[ROWS * HIDN];
  const int n0 = blockIdx.x * ROWS;
  for (int q = threadIdx.x; q < ROWS * 16; q += 128) {
    const int off = q * 8;
    *(uint4*)(sA + off) = *(const uint4*)(xf16 + (size_t)n0 * HIDN + off);
  }
  __syncthreads();
  gemm_tile(sA, HIDN, W1p, b1, true, sOut, 8, HIDN);
  __syncthreads();
  cvt_tile(sOut, sH);
  __syncthreads();
  gemm_tile(sH, HIDN, W2p, nullptr, false, sOut, 1, 16);  // N padded 2->16
  __syncthreads();
  if (threadIdx.x < ROWS * 2) {
    const int r = threadIdx.x >> 1, c = threadIdx.x & 1;
    out[(size_t)(n0 + r) * 2 + c] = sOut[r * 16 + c] + b2[c];
  }
}

// ---------------------------------------------------------------------------
// Host launcher.
// Input order (setup_inputs dict, params flattened as a pytree: dict keys in
// sorted order; mlp dicts sort as W1,W2,b1,b2,be,g):
//  0 x  1 edge_index  2 edge_attr  3 mean_vec  4 std_vec  5 mean_edge_vec
//  6 std_edge_vec  7 dec_W1  8 dec_W2  9 dec_b1  10 dec_b2
//  11-16 edge_enc{W1,W2,b1,b2,be,g}  17-22 node_enc{...}
//  23-28 proc_edge{...}  29-34 proc_node{...}
// ---------------------------------------------------------------------------
extern "C" void kernel_launch(void* const* d_in, const int* in_sizes, int n_in,
                              void* d_out, int out_size, void* d_ws, size_t ws_size,
                              hipStream_t stream) {
  const float* x       = (const float*)d_in[0];
  const int*   eidx    = (const int*)d_in[1];
  const float* eattr   = (const float*)d_in[2];
  const float* mean_v  = (const float*)d_in[3];
  const float* std_v   = (const float*)d_in[4];
  const float* mean_e  = (const float*)d_in[5];
  const float* std_e   = (const float*)d_in[6];
  const float* dec_W1  = (const float*)d_in[7];
  const float* dec_W2  = (const float*)d_in[8];
  const float* dec_b1  = (const float*)d_in[9];
  const float* dec_b2  = (const float*)d_in[10];
  const float* ee_W1 = (const float*)d_in[11];
  const float* ee_W2 = (const float*)d_in[12];
  const float* ee_b1 = (const float*)d_in[13];
  const float* ee_b2 = (const float*)d_in[14];
  const float* ee_be = (const float*)d_in[15];
  const float* ee_g  = (const float*)d_in[16];
  const float* ne_W1 = (const float*)d_in[17];
  const float* ne_W2 = (const float*)d_in[18];
  const float* ne_b1 = (const float*)d_in[19];
  const float* ne_b2 = (const float*)d_in[20];
  const float* ne_be = (const float*)d_in[21];
  const float* ne_g  = (const float*)d_in[22];
  const float* pe_W1 = (const float*)d_in[23];
  const float* pe_W2 = (const float*)d_in[24];
  const float* pe_b1 = (const float*)d_in[25];
  const float* pe_b2 = (const float*)d_in[26];
  const float* pe_be = (const float*)d_in[27];
  const float* pe_g  = (const float*)d_in[28];
  const float* pn_W1 = (const float*)d_in[29];
  const float* pn_W2 = (const float*)d_in[30];
  const float* pn_b1 = (const float*)d_in[31];
  const float* pn_b2 = (const float*)d_in[32];
  const float* pn_be = (const float*)d_in[33];
  const float* pn_g  = (const float*)d_in[34];

  const int nN = in_sizes[0] / 11;   // 100000
  const int nE = in_sizes[2] / 3;    // 700000
  const int* srcIdx = eidx;          // edge_index[0]
  const int* dstIdx = eidx + nE;     // edge_index[1]

  // Workspace carve-out (256B aligned slices).
  char*  ws  = (char*)d_ws;
  size_t off = 0;
  auto alloc = [&](size_t bytes) -> void* {
    void* p = ws + off;
    off = (off + bytes + 255) & ~(size_t)255;
    return p;
  };
  float*    xf32 = (float*)alloc((size_t)nN * HIDN * 4);
  _Float16* xf16 = (_Float16*)alloc((size_t)nN * HIDN * 2);
  float*    ef32 = (float*)alloc((size_t)nE * HIDN * 4);
  _Float16* ef16 = (_Float16*)alloc((size_t)nE * HIDN * 2);
  float*    agg  = (float*)alloc((size_t)nN * HIDN * 4);
  // Packed f16 B fragments: 1024 B per 32x16 fragment, frag count = KT*NT.
  _Float16* neW1p = (_Float16*)alloc((size_t)1  * 8 * 1024);
  _Float16* neW2p = (_Float16*)alloc((size_t)4  * 8 * 1024);
  _Float16* eeW1p = (_Float16*)alloc((size_t)1  * 8 * 1024);
  _Float16* eeW2p = (_Float16*)alloc((size_t)4  * 8 * 1024);
  _Float16* peW1p = (_Float16*)alloc((size_t)12 * 8 * 1024);
  _Float16* peW2p = (_Float16*)alloc((size_t)4  * 8 * 1024);
  _Float16* pnW1p = (_Float16*)alloc((size_t)8  * 8 * 1024);
  _Float16* pnW2p = (_Float16*)alloc((size_t)4  * 8 * 1024);
  _Float16* dW1p  = (_Float16*)alloc((size_t)4  * 8 * 1024);
  _Float16* dW2p  = (_Float16*)alloc((size_t)4  * 1 * 1024);

  // Pack all weights into WMMA B-fragment layout (K zero-padded to mult. 32).
  k_pack_w<<<8 * 1,  32, 0, stream>>>(ne_W1, 11,  32,  128, neW1p);
  k_pack_w<<<8 * 4,  32, 0, stream>>>(ne_W2, 128, 128, 128, neW2p);
  k_pack_w<<<8 * 1,  32, 0, stream>>>(ee_W1, 3,   32,  128, eeW1p);
  k_pack_w<<<8 * 4,  32, 0, stream>>>(ee_W2, 128, 128, 128, eeW2p);
  k_pack_w<<<8 * 12, 32, 0, stream>>>(pe_W1, 384, 384, 128, peW1p);
  k_pack_w<<<8 * 4,  32, 0, stream>>>(pe_W2, 128, 128, 128, peW2p);
  k_pack_w<<<8 * 8,  32, 0, stream>>>(pn_W1, 256, 256, 128, pnW1p);
  k_pack_w<<<8 * 4,  32, 0, stream>>>(pn_W2, 128, 128, 128, pnW2p);
  k_pack_w<<<8 * 4,  32, 0, stream>>>(dec_W1, 128, 128, 128, dW1p);
  k_pack_w<<<1 * 4,  32, 0, stream>>>(dec_W2, 128, 128, 2,   dW2p);

  const int gN = nN / ROWS;   // 3125 (exact)
  const int gE = nE / ROWS;   // 21875 (exact)

  k_node_enc<<<gN, 128, 0, stream>>>(x, mean_v, std_v, neW1p, ne_b1, neW2p, ne_b2,
                                     ne_g, ne_be, xf32, xf16, agg);
  k_edge_enc<<<gE, 128, 0, stream>>>(eattr, mean_e, std_e, eeW1p, ee_b1, eeW2p, ee_b2,
                                     ee_g, ee_be, ef32, ef16);

  for (int l = 0; l < 15; ++l) {   // shared weights across layers
    k_edge_proc<<<gE, 128, 0, stream>>>(srcIdx, dstIdx, xf16, ef32, ef16,
                                        peW1p, pe_b1, peW2p, pe_b2, pe_g, pe_be, agg);
    k_node_proc<<<gN, 128, 0, stream>>>(xf32, xf16, agg,
                                        pnW1p, pn_b1, pnW2p, pn_b2, pn_g, pn_be);
  }

  k_decoder<<<gN, 128, 0, stream>>>(xf16, dW1p, dec_b1, dW2p, dec_b2, (float*)d_out);
}
